// MultiHeadAttention_64183991271479
// MI455X (gfx1250) — matrix operand order repaired and checked by reading, not compiled, not run on hardware
//
#include <hip/hip_runtime.h>
#include <cstdint>

// -------- MHA: B=4, T=2048, D=1024, H=16 (d_k = 64) ----------------------
// Pipeline: 3x GEMM (x@W+b -> Q/K/V, f32->f16), flash-attention (WMMA f16,
// f32 accum, online softmax), GEMM (ctx@Wo+bo -> f32 out).
// All matmuls use v_wmma_f32_16x16x32_f16.  K tiles in the attention kernel
// are staged with the Tensor Data Mover (tensor_load_to_lds + TENSORcnt)
// when the builtin is available; next-tile global prefetches warm GL2.

typedef __attribute__((ext_vector_type(16))) _Float16 v16h;
typedef __attribute__((ext_vector_type(8)))  _Float16 v8h;
typedef __attribute__((ext_vector_type(4)))  _Float16 v4h;
typedef __attribute__((ext_vector_type(8)))  float    v8f;
typedef __attribute__((ext_vector_type(4)))  float    v4f;
typedef __attribute__((ext_vector_type(4)))  unsigned int u32x4;
typedef __attribute__((ext_vector_type(8)))  int      i32x8;
typedef __attribute__((ext_vector_type(4)))  int      i32x4;

#define LDSP 72   // padded LDS row stride in f16 elements (144 B, 16B aligned)

#if defined(__has_builtin)
#  if __has_builtin(__builtin_amdgcn_tensor_load_to_lds)
#    define HAVE_TDM 1
#  endif
#endif
#ifndef HAVE_TDM
#  define HAVE_TDM 0
#endif

// ---- WMMA fragment loaders (ISA 7.12.2 layouts, wave32) ------------------
// A-matrix 16x32 f16 fragment: lane L -> row m = L&15, khalf = (L>>4)*8,
// elems 0..7 -> k = khalf+e ; elems 8..15 -> k = 16+khalf+(e-8).
__device__ inline v16h load_a_frag(const _Float16* base, int ld, int lane) {
    const _Float16* row = base + (size_t)(lane & 15) * ld + ((lane >> 4) * 8);
    union { v16h v; v8h h[2]; } u;
    u.h[0] = *(const v8h*)(row);
    u.h[1] = *(const v8h*)(row + 16);
    return u.v;
}
// B-matrix 32x16 f16 fragment from K-contiguous src[n][k]:
// lane L -> col n = L&15, khalf = (L>>4)*8, elems 0..7 -> k = khalf+e ;
// elems 8..15 -> k = 16+khalf+(e-8).
__device__ inline v16h load_b_frag(const _Float16* base, int ld, int lane) {
    const _Float16* row = base + (size_t)(lane & 15) * ld + ((lane >> 4) * 8);
    union { v16h v; v8h h[2]; } u;
    u.h[0] = *(const v8h*)(row);
    u.h[1] = *(const v8h*)(row + 16);
    return u.v;
}

// ---- dtype helpers --------------------------------------------------------
__device__ inline void load4h(const float* p, _Float16* o) {
    v4f t = *(const v4f*)p;
    o[0] = (_Float16)t.x; o[1] = (_Float16)t.y;
    o[2] = (_Float16)t.z; o[3] = (_Float16)t.w;
}
__device__ inline void load4h(const _Float16* p, _Float16* o) {
    v4h t = *(const v4h*)p;
    o[0] = t.x; o[1] = t.y; o[2] = t.z; o[3] = t.w;
}

// ---- GEMM: C[M,N] = A[M,K] @ W[K,N] + bias, f16 WMMA, f32 accum ----------
// Block tile 64(M) x 128(N), 4 waves; wave tile 64x32.  K staged in steps of
// 64 with register double-buffering: global loads for step k+1 issue while
// the WMMAs for step k run out of LDS.
template <typename AT>
__device__ __attribute__((always_inline)) inline
void gemm_load_regs(const AT* __restrict__ A, const float* __restrict__ W,
                    int N, int K, int m0, int n0, int k0, int tid,
                    _Float16 aReg[8][4], _Float16 wReg[4][4][4]) {
    #pragma unroll
    for (int i = 0; i < 8; ++i) {                 // A: 64x64, vec4 chunks
        int c = tid + i * 128;
        int row = c >> 4, kk = (c & 15) * 4;
        load4h(&A[(size_t)(m0 + row) * K + k0 + kk], aReg[i]);
    }
    #pragma unroll
    for (int i = 0; i < 4; ++i) {                 // W: 64x128 as 4x4 micro-tiles
        int mt = tid + i * 128;
        int kr = (mt >> 5) * 4, nn = (mt & 31) * 4;
        #pragma unroll
        for (int j = 0; j < 4; ++j)
            load4h(&W[(size_t)(k0 + kr + j) * N + n0 + nn], wReg[i][j]);
    }
}

__device__ __attribute__((always_inline)) inline
void gemm_store_lds(_Float16 (*As)[LDSP], _Float16 (*Wt)[LDSP], int tid,
                    const _Float16 aReg[8][4], const _Float16 wReg[4][4][4]) {
    #pragma unroll
    for (int i = 0; i < 8; ++i) {
        int c = tid + i * 128;
        int row = c >> 4, kk = (c & 15) * 4;
        v4h t; t.x = aReg[i][0]; t.y = aReg[i][1]; t.z = aReg[i][2]; t.w = aReg[i][3];
        *(v4h*)(&As[row][kk]) = t;
    }
    #pragma unroll
    for (int i = 0; i < 4; ++i) {                 // transposed 4x4 -> ds_store_b64
        int mt = tid + i * 128;
        int kr = (mt >> 5) * 4, nn = (mt & 31) * 4;
        #pragma unroll
        for (int jj = 0; jj < 4; ++jj) {
            v4h t; t.x = wReg[i][0][jj]; t.y = wReg[i][1][jj];
                   t.z = wReg[i][2][jj]; t.w = wReg[i][3][jj];
            *(v4h*)(&Wt[nn + jj][kr]) = t;
        }
    }
}

template <typename AT, typename OT>
__global__ __launch_bounds__(128)
void gemm_bias_kernel(const AT* __restrict__ A, const float* __restrict__ W,
                      const float* __restrict__ bias, OT* __restrict__ C,
                      int M, int N, int K) {
    __shared__ __align__(16) _Float16 As[64][LDSP];   // (m, k)
    __shared__ __align__(16) _Float16 Wt[128][LDSP];  // (n, k)  transposed

    const int tid  = threadIdx.x;
    const int lane = tid & 31;
    const int wave = tid >> 5;
    const int lcol = lane & 15;
    const int lhalf = lane >> 4;
    const int m0 = blockIdx.y * 64;
    const int n0 = blockIdx.x * 128;

    v8f acc[4][2] = {};
    _Float16 aReg[8][4];
    _Float16 wReg[4][4][4];

    gemm_load_regs(A, W, N, K, m0, n0, 0, tid, aReg, wReg);

    for (int k0 = 0; k0 < K; k0 += 64) {
        gemm_store_lds(As, Wt, tid, aReg, wReg);
        __syncthreads();

        if (k0 + 64 < K)                              // overlap with compute
            gemm_load_regs(A, W, N, K, m0, n0, k0 + 64, tid, aReg, wReg);
        if (k0 + 128 < K) {                           // warm GL2 one step ahead
            int row = tid >> 1, half = (tid & 1) * 32;
            __builtin_prefetch(&A[(size_t)(m0 + row) * K + k0 + 128 + half], 0, 1);
            __builtin_prefetch(&W[(size_t)(k0 + 128 + row) * N + n0 + half * 2], 0, 1);
        }

        #pragma unroll
        for (int kc = 0; kc < 2; ++kc) {
            v16h a[4], b[2];
            #pragma unroll
            for (int mi = 0; mi < 4; ++mi)
                a[mi] = load_a_frag(&As[mi * 16][0] + kc * 32, LDSP, lane);
            #pragma unroll
            for (int ni = 0; ni < 2; ++ni)
                b[ni] = load_b_frag(&Wt[wave * 32 + ni * 16][0] + kc * 32, LDSP, lane);
            #pragma unroll
            for (int mi = 0; mi < 4; ++mi)
                #pragma unroll
                for (int ni = 0; ni < 2; ++ni)
                    acc[mi][ni] = __builtin_amdgcn_wmma_f32_16x16x32_f16(
                        false, a[mi], false, b[ni], (short)0, acc[mi][ni],
                        false, false);
        }
        __syncthreads();
    }

    // epilogue: + bias, store (C/D layout: m=(lane>>4)*8+r, n=lane&15)
    #pragma unroll
    for (int ni = 0; ni < 2; ++ni) {
        int n = n0 + wave * 32 + ni * 16 + lcol;
        float bv = bias[n];
        #pragma unroll
        for (int mi = 0; mi < 4; ++mi)
            #pragma unroll
            for (int r = 0; r < 8; ++r) {
                int m = m0 + mi * 16 + lhalf * 8 + r;
                C[(size_t)m * N + n] = (OT)(acc[mi][ni][r] + bv);
            }
    }
}

// ---- Flash attention: causal, per (batch, head, 64-query tile) -----------
__global__ __launch_bounds__(128)
void attn_kernel(const _Float16* __restrict__ Q, const _Float16* __restrict__ K,
                 const _Float16* __restrict__ V, _Float16* __restrict__ O,
                 int T, int D) {
    __shared__ __align__(16) _Float16 Ks[64][LDSP];       // (key, d)
    __shared__ __align__(16) _Float16 Vt[64][LDSP];       // (dv, key)
    __shared__ __align__(16) _Float16 Ps[4][16][LDSP];    // per-wave P (q, key)

    const int tid  = threadIdx.x;
    const int lane = tid & 31;
    const int wave = tid >> 5;
    const int lcol = lane & 15;
    const int lhalf = lane >> 4;
    const int b  = blockIdx.z;
    const int h  = blockIdx.y;
    const int q0 = blockIdx.x * 64;

    // Q fragments for this wave's 16 rows (two K=32 chunks over d_k=64)
    const _Float16* Qbase = Q + ((size_t)b * T + q0 + wave * 16) * D + h * 64;
    v16h qf[2];
    #pragma unroll
    for (int kc = 0; kc < 2; ++kc)
        qf[kc] = load_a_frag(Qbase + kc * 32, D, lane);

    v8f o_acc[4] = {};
    float m_i[8], l_i[8];
    #pragma unroll
    for (int r = 0; r < 8; ++r) { m_i[r] = -3.0e38f; l_i[r] = 0.0f; }

    const int ktiles = q0 / 64 + 1;
    for (int kt = 0; kt < ktiles; ++kt) {
        const int kbase = kt * 64;

#if HAVE_TDM
        // --- K tile via Tensor Data Mover: 64x64 f16, row stride D elems,
        //     LDS pad 4 DWORDs every 32 DWORDs -> 144 B rows (= LDSP).
        if (wave == 0) {
            uint64_t ga = (uint64_t)(uintptr_t)(K + ((size_t)b * T + kbase) * D + h * 64);
            uint32_t la = (uint32_t)(uintptr_t)
                (__attribute__((address_space(3))) const void*)&Ks[0][0];
            u32x4 g0;
            g0.x = 1u;                                   // count=1 (valid user D#)
            g0.y = la;                                   // lds_addr
            g0.z = (uint32_t)ga;                         // global_addr[31:0]
            g0.w = (uint32_t)((ga >> 32) & 0x01FFFFFFu)  // global_addr[56:32]
                   | (2u << 30);                         // type = 2 ("image")
            i32x8 g1;
            g1[0] = (1 << 16)      // data_size = 1 -> 2 bytes
                  | (1 << 20)      // pad_enable
                  | (4 << 22)      // pad_interval: 32 DWORDs (128 B)
                  | (3 << 25);     // pad_amount: 4 DWORDs (16 B)
            g1[1] = (int)(64u << 16);   // tensor_dim0 = 64   (bits 79:48)
            g1[2] = (int)(64u << 16);   // tensor_dim1 = 64   (bits 111:80)
            g1[3] = (int)(64u << 16);   // tile_dim0  = 64    (bits 127:112)
            g1[4] = 64;                 // tile_dim1  = 64    (bits 143:128)
            g1[5] = 1024;               // tensor_dim0_stride = D (bits 207:160)
            g1[6] = 0;
            g1[7] = 0;
            i32x4 gz = {0, 0, 0, 0};    // groups 2/3 unused for 2D tiles
#if __clang_major__ >= 23
            i32x8 gz8 = {0, 0, 0, 0, 0, 0, 0, 0};
            __builtin_amdgcn_tensor_load_to_lds(g0, g1, gz, gz, gz8, 0);
#else
            __builtin_amdgcn_tensor_load_to_lds(g0, g1, gz, gz, 0);
#endif
#if __has_builtin(__builtin_amdgcn_s_wait_tensorcnt)
            __builtin_amdgcn_s_wait_tensorcnt((short)0);
#else
            asm volatile("s_wait_tensorcnt 0x0" ::: "memory");
#endif
        }
#else
        // --- K tile staged by all threads (fallback)
        #pragma unroll
        for (int i = 0; i < 4; ++i) {
            int c = tid + i * 128;
            int row = c >> 3, dd = (c & 7) * 8;
            *(v8h*)(&Ks[row][dd]) =
                *(const v8h*)(K + ((size_t)b * T + kbase + row) * D + h * 64 + dd);
        }
#endif
        // --- V tile transposed: 4(key) x 8(dv) micro-tiles -> ds_store_b64
        {
            int key4 = (tid & 15) * 4, dv8 = (tid >> 4) * 8;
            v8h vv[4];
            #pragma unroll
            for (int j = 0; j < 4; ++j)
                vv[j] = *(const v8h*)(V + ((size_t)b * T + kbase + key4 + j) * D
                                        + h * 64 + dv8);
            #pragma unroll
            for (int jj = 0; jj < 8; ++jj) {
                v4h t; t.x = vv[0][jj]; t.y = vv[1][jj];
                       t.z = vv[2][jj]; t.w = vv[3][jj];
                *(v4h*)(&Vt[dv8 + jj][key4]) = t;
            }
        }
        if (kt + 1 < ktiles) {                       // warm GL2 for next tile
            int row = tid >> 1, half = (tid & 1) * 32;
            size_t g = ((size_t)b * T + kbase + 64 + row) * D + h * 64 + half;
            __builtin_prefetch(K + g, 0, 1);
            __builtin_prefetch(V + g, 0, 1);
        }
        __syncthreads();

        // S = Q K^T  (4 key-chunks of 16; reduce over d_k = 2 x K=32)
        v8f s[4];
        #pragma unroll
        for (int ns = 0; ns < 4; ++ns) {
            v8f z = {};
            #pragma unroll
            for (int kc = 0; kc < 2; ++kc) {
                v16h bf = load_b_frag(&Ks[ns * 16][0] + kc * 32, LDSP, lane);
                z = __builtin_amdgcn_wmma_f32_16x16x32_f16(
                        false, qf[kc], false, bf, (short)0, z, false, false);
            }
            s[ns] = z;
        }

        // scale + causal mask (diagonal tile only)
        const bool diag = (kbase == q0);
        #pragma unroll
        for (int ns = 0; ns < 4; ++ns)
            #pragma unroll
            for (int r = 0; r < 8; ++r) {
                float x = s[ns][r] * 0.125f;          // 1/sqrt(64)
                if (diag) {
                    int key  = kbase + ns * 16 + lcol;
                    int qrow = q0 + wave * 16 + lhalf * 8 + r;
                    if (key > qrow) x = -3.0e38f;
                }
                s[ns][r] = x;
            }

        // online softmax (rows m = lhalf*8 + r; reduce across 16-lane half)
        #pragma unroll
        for (int r = 0; r < 8; ++r) {
            float mx = fmaxf(fmaxf(s[0][r], s[1][r]), fmaxf(s[2][r], s[3][r]));
            #pragma unroll
            for (int off = 1; off < 16; off <<= 1)
                mx = fmaxf(mx, __shfl_xor(mx, off, 32));
            float nm   = fmaxf(m_i[r], mx);
            float corr = __expf(m_i[r] - nm);
            float sum = 0.0f;
            #pragma unroll
            for (int ns = 0; ns < 4; ++ns) {
                float p = __expf(s[ns][r] - nm);
                s[ns][r] = p;
                sum += p;
            }
            #pragma unroll
            for (int off = 1; off < 16; off <<= 1)
                sum += __shfl_xor(sum, off, 32);
            l_i[r] = l_i[r] * corr + sum;
            m_i[r] = nm;
            #pragma unroll
            for (int nc = 0; nc < 4; ++nc) o_acc[nc][r] *= corr;
        }

        // P -> LDS (f16), then O += P @ V via WMMA
        #pragma unroll
        for (int ns = 0; ns < 4; ++ns)
            #pragma unroll
            for (int r = 0; r < 8; ++r)
                Ps[wave][lhalf * 8 + r][ns * 16 + lcol] = (_Float16)s[ns][r];
        __syncthreads();

        #pragma unroll
        for (int kc = 0; kc < 2; ++kc) {
            v16h pa = load_a_frag(&Ps[wave][0][0] + kc * 32, LDSP, lane);
            #pragma unroll
            for (int nc = 0; nc < 4; ++nc) {
                v16h vb = load_b_frag(&Vt[nc * 16][0] + kc * 32, LDSP, lane);
                o_acc[nc] = __builtin_amdgcn_wmma_f32_16x16x32_f16(
                        false, pa, false, vb, (short)0, o_acc[nc], false, false);
            }
        }
        __syncthreads();   // Ks/Vt reused next iteration
    }

    // normalize and store ctx in (b, t, h, dv) f16 layout
    _Float16* Obase = O + ((size_t)b * T + q0 + wave * 16) * D + h * 64;
    #pragma unroll
    for (int r = 0; r < 8; ++r) {
        float inv = 1.0f / l_i[r];
        int m = lhalf * 8 + r;
        #pragma unroll
        for (int nc = 0; nc < 4; ++nc) {
            int dv = nc * 16 + lcol;
            Obase[(size_t)m * D + dv] = (_Float16)(o_acc[nc][r] * inv);
        }
    }
}

// ---- host launch ----------------------------------------------------------
extern "C" void kernel_launch(void* const* d_in, const int* in_sizes, int n_in,
                              void* d_out, int out_size, void* d_ws, size_t ws_size,
                              hipStream_t stream) {
    const float* x  = (const float*)d_in[0];
    const float* Wq = (const float*)d_in[1];
    const float* bq = (const float*)d_in[2];
    const float* Wk = (const float*)d_in[3];
    const float* bk = (const float*)d_in[4];
    const float* Wv = (const float*)d_in[5];
    const float* bv = (const float*)d_in[6];
    const float* Wo = (const float*)d_in[7];
    const float* bo = (const float*)d_in[8];
    float* out = (float*)d_out;

    const int B = 4, T = 2048, D = 1024, H = 16;
    const int M = B * T;                       // 8192

    _Float16* Qh = (_Float16*)d_ws;            // f16 intermediates in scratch
    _Float16* Kh = Qh + (size_t)M * D;
    _Float16* Vh = Kh + (size_t)M * D;
    _Float16* Ch = Vh + (size_t)M * D;

    dim3 block(128);
    dim3 ggrid(D / 128, M / 64);               // (8, 128)

    gemm_bias_kernel<float, _Float16><<<ggrid, block, 0, stream>>>(x, Wq, bq, Qh, M, D, D);
    gemm_bias_kernel<float, _Float16><<<ggrid, block, 0, stream>>>(x, Wk, bk, Kh, M, D, D);
    gemm_bias_kernel<float, _Float16><<<ggrid, block, 0, stream>>>(x, Wv, bv, Vh, M, D, D);

    dim3 agrid(T / 64, H, B);                  // (32, 16, 4)
    attn_kernel<<<agrid, block, 0, stream>>>(Qh, Kh, Vh, Ch, T, D);

    gemm_bias_kernel<_Float16, float><<<ggrid, block, 0, stream>>>(Ch, Wo, bo, out, M, D, D);
}
